// GCNFusionBlock_54065048322217
// MI455X (gfx1250) — matrix operand order repaired and checked by reading, not compiled, not run on hardware
//
#include <hip/hip_runtime.h>

typedef _Float16 v16h __attribute__((ext_vector_type(16)));
typedef float v8f __attribute__((ext_vector_type(8)));

#define NB   8
#define SEQ  4096
#define DIM  256
#define NH   4
#define TOPP 256
#define TOPN 128
#define NQT  384   // TOPP + TOPN

// ---------------------------------------------------------------------------
// WMMA fragment helpers (wave32, v_wmma_f32_16x16x32_f16)
// A 16x32 (MxK): lane<16 -> M=lane, k = {kb..kb+7, kb+16..kb+23}, kb=0 (kb=8 for
// lanes 16-31). Both runs are CONTIGUOUS 16B -> two b128 loads per lane.
// B 32x16 (KxN): lane<16 -> N=lane, k = {kb..kb+7, kb+16..kb+23} (same as A).
// With B stored as [N][K] row-major this is two contiguous 16B runs per lane.
// C/D: VGPR j: lanes 0-15 -> M=j, N=lane; lanes 16-31 -> M=j+8, N=lane-16.
// ---------------------------------------------------------------------------
__device__ inline v16h load2x8h(const _Float16* p0, const _Float16* p1) {
  union { uint4 u[2]; v16h h; } u;
  u.u[0] = *(const uint4*)p0;
  u.u[1] = *(const uint4*)p1;
  return u.h;
}

__device__ inline v8f vzero8() {
  v8f z = {0.f, 0.f, 0.f, 0.f, 0.f, 0.f, 0.f, 0.f};
  return z;
}

// ---------------------------------------------------------------------------
// LDS-free batched WMMA GEMM: C[M,N] = A[M,K] @ B, with B stored as Bt[N][K]
// row-major (ldb = K stride). fp16 in, fp32 accumulate. Optional relu, fp32
// and/or fp16 output, optional transposed fp16 output (C16t[N][M]).
// Block = 4 waves; each wave computes 32 rows x 64 cols. M%128==0, N%64==0,
// K%32==0 at every call site.
// ---------------------------------------------------------------------------
struct GemmP {
  const _Float16* A;
  const _Float16* Bt;
  float* C32;
  _Float16* C16;
  long long sA, sB, sC;
  int M, N, K, lda, ldb, transC16, relu;
};

__global__ __launch_bounds__(128) void gemm_wmma(GemmP p) {
  int bb = blockIdx.z;
  const _Float16* A = p.A + (long long)bb * p.sA;
  const _Float16* Bt = p.Bt + (long long)bb * p.sB;
  int m0 = blockIdx.x * 128, n0 = blockIdx.y * 64;
  int lane = threadIdx.x & 31, w = threadIdx.x >> 5;
  int l16 = lane & 15;
  int kb = (lane < 16) ? 0 : 8;
  int ks = (lane < 16) ? 0 : 16;

  const _Float16* arow0 = A + (long long)(m0 + w * 32 + l16) * p.lda;
  const _Float16* arow1 = arow0 + (long long)16 * p.lda;

  v8f acc[2][4];
#pragma unroll
  for (int s = 0; s < 2; ++s)
#pragma unroll
    for (int c = 0; c < 4; ++c) acc[s][c] = vzero8();

  for (int k0 = 0; k0 < p.K; k0 += 32) {
    v16h a0 = load2x8h(arow0 + k0 + kb, arow0 + k0 + kb + 16);
    v16h a1 = load2x8h(arow1 + k0 + kb, arow1 + k0 + kb + 16);
#pragma unroll
    for (int c = 0; c < 4; ++c) {
      const _Float16* brow = Bt + (long long)(n0 + c * 16 + l16) * p.ldb + k0 + kb;
      v16h bf = load2x8h(brow, brow + 16);
      acc[0][c] = __builtin_amdgcn_wmma_f32_16x16x32_f16(false, a0, false, bf,
                                                         (short)0, acc[0][c], false, false);
      acc[1][c] = __builtin_amdgcn_wmma_f32_16x16x32_f16(false, a1, false, bf,
                                                         (short)0, acc[1][c], false, false);
    }
  }

  int cb = l16;
#pragma unroll
  for (int s = 0; s < 2; ++s) {
    int rbase = m0 + w * 32 + s * 16 + ((lane < 16) ? 0 : 8);
#pragma unroll
    for (int c = 0; c < 4; ++c) {
#pragma unroll
      for (int j = 0; j < 8; ++j) {
        int row = rbase + j;
        int col = n0 + c * 16 + cb;
        float v = acc[s][c][j];
        if (p.relu) v = v > 0.f ? v : 0.f;
        if (p.C32) p.C32[(long long)bb * p.sC + (long long)row * p.N + col] = v;
        if (p.C16) {
          long long idx = p.transC16 ? ((long long)bb * p.sC + (long long)col * p.M + row)
                                     : ((long long)bb * p.sC + (long long)row * p.N + col);
          p.C16[idx] = (_Float16)v;
        }
      }
    }
  }
}

// ---------------------------------------------------------------------------
// Flash attention, one wave per (16-query tile, head, batch), dh = 64.
// Q [nq,256] row-major, K [nk,256] row-major, V TRANSPOSED: Vt [256, nk]
// (ldv = nk). Out fp16 [nq,256]. All K/V/Q fragments are direct b128 global
// loads; only the 16x32 P tile goes through LDS (cross-lane transpose).
// ---------------------------------------------------------------------------
__global__ __launch_bounds__(32) void attn_kernel(const _Float16* Q, const _Float16* Kg,
                                                  const _Float16* Vt, _Float16* O,
                                                  int nq, int nk, int ldv,
                                                  long long sQ, long long sK,
                                                  long long sV, long long sO,
                                                  float scale) {
  __shared__ _Float16 Ps[16 * 48];
  int lane = threadIdx.x;
  int l16 = lane & 15;
  int kb = (lane < 16) ? 0 : 8;
  int ks = (lane < 16) ? 0 : 16;
  int q0 = blockIdx.x * 16, h = blockIdx.y, bb = blockIdx.z;

  const _Float16* Kbp = Kg + (long long)bb * sK + h * 64;
  const _Float16* Vbp = Vt + (long long)bb * sV;
  _Float16* Op = O + (long long)bb * sO + h * 64;

  // preload Q fragments (dh=64 -> two 32-deep K chunks)
  const _Float16* qrow = Q + (long long)bb * sQ + (long long)(q0 + l16) * DIM + h * 64;
  v16h aq0 = load2x8h(qrow + kb, qrow + kb + 16);
  v16h aq1 = load2x8h(qrow + 32 + kb, qrow + 32 + kb + 16);

  v8f o[4];
  float m8[8], l8[8];
#pragma unroll
  for (int c = 0; c < 4; ++c) o[c] = vzero8();
#pragma unroll
  for (int j = 0; j < 8; ++j) { m8[j] = -1e30f; l8[j] = 0.f; }

  for (int j0 = 0; j0 < nk; j0 += 32) {
    // S = Q @ K^T for 32 keys (two 16-col tiles)
    const _Float16* kr0 = Kbp + (long long)(j0 + l16) * DIM;
    const _Float16* kr1 = Kbp + (long long)(j0 + 16 + l16) * DIM;
    v8f s0 = vzero8(), s1 = vzero8();
    {
      v16h b = load2x8h(kr0 + kb, kr0 + kb + 16);
      s0 = __builtin_amdgcn_wmma_f32_16x16x32_f16(false, aq0, false, b, (short)0, s0, false, false);
      b = load2x8h(kr0 + 32 + kb, kr0 + 32 + kb + 16);
      s0 = __builtin_amdgcn_wmma_f32_16x16x32_f16(false, aq1, false, b, (short)0, s0, false, false);
      b = load2x8h(kr1 + kb, kr1 + kb + 16);
      s1 = __builtin_amdgcn_wmma_f32_16x16x32_f16(false, aq0, false, b, (short)0, s1, false, false);
      b = load2x8h(kr1 + 32 + kb, kr1 + 32 + kb + 16);
      s1 = __builtin_amdgcn_wmma_f32_16x16x32_f16(false, aq1, false, b, (short)0, s1, false, false);
    }

    float corr8[8];
#pragma unroll
    for (int j = 0; j < 8; ++j) {
      float v0 = s0[j] * scale, v1 = s1[j] * scale;
      float t = fmaxf(v0, v1);
      t = fmaxf(t, __shfl_xor(t, 1)); t = fmaxf(t, __shfl_xor(t, 2));
      t = fmaxf(t, __shfl_xor(t, 4)); t = fmaxf(t, __shfl_xor(t, 8));
      float mn = fmaxf(m8[j], t);
      float corr = __expf(m8[j] - mn);
      float e0 = __expf(v0 - mn), e1 = __expf(v1 - mn);
      float r = e0 + e1;
      r += __shfl_xor(r, 1); r += __shfl_xor(r, 2);
      r += __shfl_xor(r, 4); r += __shfl_xor(r, 8);
      l8[j] = l8[j] * corr + r;
      m8[j] = mn;
      corr8[j] = corr;
      int prow = (lane < 16) ? j : (j + 8);
      Ps[prow * 48 + l16] = (_Float16)e0;
      Ps[prow * 48 + 16 + l16] = (_Float16)e1;
    }
#pragma unroll
    for (int c = 0; c < 4; ++c)
#pragma unroll
      for (int j = 0; j < 8; ++j) o[c][j] *= corr8[j];
    __syncthreads();

    // O += P @ V   (V fragments contiguous from Vt[dh][key])
    v16h pa = load2x8h(&Ps[l16 * 48 + kb], &Ps[l16 * 48 + kb + 16]);
#pragma unroll
    for (int c = 0; c < 4; ++c) {
      const _Float16* vr = Vbp + (long long)(h * 64 + c * 16 + l16) * ldv + j0 + kb;
      v16h bv = load2x8h(vr, vr + 16);
      o[c] = __builtin_amdgcn_wmma_f32_16x16x32_f16(false, pa, false, bv, (short)0, o[c], false, false);
    }
    __syncthreads();
  }

#pragma unroll
  for (int j = 0; j < 8; ++j) l8[j] = 1.f / l8[j];
  int rb = (lane < 16) ? 0 : 8;
#pragma unroll
  for (int c = 0; c < 4; ++c)
#pragma unroll
    for (int j = 0; j < 8; ++j)
      Op[(long long)(q0 + rb + j) * DIM + c * 16 + l16] = (_Float16)(o[c][j] * l8[j]);
}

// ---------------------------------------------------------------------------
// Small elementwise / reduction kernels
// ---------------------------------------------------------------------------
__global__ void f2h_t_kernel(const float* s, _Float16* d) {  // d[c*DIM+r] = s[r*DIM+c]
  int i = blockIdx.x * 256 + threadIdx.x;
  int r = i >> 8, c = i & 255;
  d[c * DIM + r] = (_Float16)s[i];
}

__global__ void masked_mean_kernel(const float* x, const int* mask, float* pq, float* nq) {
  int b = blockIdx.x, d = threadIdx.x;
  float sp = 0.f, sn = 0.f, cp = 0.f, cn = 0.f;
  for (int n = 0; n < SEQ; ++n) {
    int m = mask[b * SEQ + n];
    float v = x[((long long)b * SEQ + n) * DIM + d];
    if (m == 1) { sp += v; cp += 1.f; }
    else if (m == -1) { sn += v; cn += 1.f; }
  }
  pq[b * DIM + d] = sp / (cp + 1e-6f);
  nq[b * DIM + d] = sn / (cn + 1e-6f);
}

// t[b] = W q[b];  c[b] = bias . q[b]   (fold (x@W+b).q == x.(Wq) + b.q)
__global__ void simvec_kernel(const float* W, const float* bias, const float* q,
                              float* t, float* cout) {
  __shared__ float qs[DIM];
  int b = blockIdx.x, i = threadIdx.x;
  qs[i] = q[b * DIM + i];
  __syncthreads();
  float acc = 0.f;
  const float* wr = W + (long long)i * DIM;
  for (int j = 0; j < DIM; ++j) acc += wr[j] * qs[j];
  t[b * DIM + i] = acc;
  if (i == 0) {
    float c = 0.f;
    for (int j = 0; j < DIM; ++j) c += bias[j] * qs[j];
    cout[b] = c;
  }
}

__global__ void sims_kernel(const float* x, const float* tp, const float* cp,
                            const float* tn, const float* cn,
                            float* psim_out, float* nsim_out,
                            float* psim_ws, float* nsim_ws) {
  int wave = threadIdx.x >> 5, lane = threadIdx.x & 31;
  long long row = (long long)blockIdx.x * 8 + wave;  // b*SEQ + n
  int b = (int)(row / SEQ);
  const float* xr = x + row * DIM;
  const float* tpr = tp + b * DIM;
  const float* tnr = tn + b * DIM;
  float ap = 0.f, an = 0.f;
  for (int j = lane; j < DIM; j += 32) { float v = xr[j]; ap += v * tpr[j]; an += v * tnr[j]; }
  for (int off = 16; off; off >>= 1) { ap += __shfl_xor(ap, off); an += __shfl_xor(an, off); }
  if (lane == 0) {
    float ps = 1.f / (1.f + __expf(-(ap + cp[b])));
    float ns = 1.f / (1.f + __expf(-(an + cn[b])));
    psim_out[row] = ps; nsim_out[row] = ns;
    psim_ws[row] = ps; nsim_ws[row] = ns;
  }
}

__global__ void topk_kernel(const float* sims, int* idx, float* val, int Ksel) {
  __shared__ float sv[SEQ];
  __shared__ float rv[256];
  __shared__ int ri[256];
  int b = blockIdx.x, t = threadIdx.x;
  for (int i = t; i < SEQ; i += 256) sv[i] = sims[(long long)b * SEQ + i];
  __syncthreads();
  for (int k = 0; k < Ksel; ++k) {
    float bv = -1e30f; int bi = SEQ;
    for (int i = t; i < SEQ; i += 256) {
      float v = sv[i];
      if (v > bv) { bv = v; bi = i; }
    }
    rv[t] = bv; ri[t] = bi;
    __syncthreads();
    for (int s = 128; s > 0; s >>= 1) {
      if (t < s) {
        float ov = rv[t + s]; int oi = ri[t + s];
        if (ov > rv[t] || (ov == rv[t] && oi < ri[t])) { rv[t] = ov; ri[t] = oi; }
      }
      __syncthreads();
    }
    if (t == 0) { idx[b * Ksel + k] = ri[0]; val[b * Ksel + k] = rv[0]; sv[ri[0]] = -1e30f; }
    __syncthreads();
  }
}

__global__ void gather_kernel(const float* x, const int* idx, const float* val,
                              _Float16* out, int Ksel) {
  int r = blockIdx.x;
  int b = r / Ksel;
  int d = threadIdx.x;
  int id = idx[r];
  float keep = val[r] > 0.6f ? 1.f : 0.f;
  out[(long long)r * DIM + d] = (_Float16)(x[((long long)b * SEQ + id) * DIM + d] * keep);
}

__global__ void softmax_kernel(const float* logits, _Float16* out, int cols, float scale) {
  __shared__ float red[256];
  long long row = blockIdx.x;
  int t = threadIdx.x;
  float v = (t < cols) ? logits[row * cols + t] * scale : -1e30f;
  red[t] = v; __syncthreads();
  for (int s = 128; s; s >>= 1) { if (t < s) red[t] = fmaxf(red[t], red[t + s]); __syncthreads(); }
  float mx = red[0];
  __syncthreads();
  float e = (t < cols) ? __expf(v - mx) : 0.f;
  red[t] = e; __syncthreads();
  for (int s = 128; s; s >>= 1) { if (t < s) red[t] += red[t + s]; __syncthreads(); }
  float inv = 1.f / red[0];
  if (t < cols) out[row * cols + t] = (_Float16)(e * inv);
}

__global__ void build_query_kernel(const float* qgp, const float* qgn,
                                   const int* idxp, const int* idxn,
                                   const float* pe, float* query) {
  int r = blockIdx.x;
  int b = r / NQT, slot = r % NQT, d = threadIdx.x;
  float g; int id;
  if (slot < TOPP) { g = qgp[((long long)b * TOPP + slot) * DIM + d]; id = idxp[b * TOPP + slot]; }
  else { int s = slot - TOPP; g = qgn[((long long)b * TOPN + s) * DIM + d]; id = idxn[b * TOPN + s]; }
  query[(long long)r * DIM + d] = g + pe[(long long)id * DIM + d];
}

__global__ void ln_kernel(const float* src, const float* g, const float* bta, _Float16* dst) {
  __shared__ float r1[256];
  __shared__ float r2[256];
  long long row = blockIdx.x;
  int t = threadIdx.x;
  float v = src[row * DIM + t];
  r1[t] = v; r2[t] = v * v;
  __syncthreads();
  for (int s = 128; s; s >>= 1) { if (t < s) { r1[t] += r1[t + s]; r2[t] += r2[t + s]; } __syncthreads(); }
  float mean = r1[0] * (1.f / DIM);
  float var = r2[0] * (1.f / DIM) - mean * mean;
  float inv = rsqrtf(var + 1e-5f);
  dst[row * DIM + t] = (_Float16)((v - mean) * inv * g[t] + bta[t]);
}

__global__ void gate_res_kernel(const float* res, const float* c, const float* bias,
                                const float* gate, float* dst) {
  long long i = (long long)blockIdx.x * DIM + threadIdx.x;
  int t = threadIdx.x;
  dst[i] = res[i] + gate[t] * (c[i] + bias[t]);
}

__global__ void final_add_kernel(float* dst, const float* m, const float* bias) {
  long long i = (long long)blockIdx.x * DIM + threadIdx.x;
  dst[i] += m[i] + bias[threadIdx.x];
}

// ---------------------------------------------------------------------------
extern "C" void kernel_launch(void* const* d_in, const int* in_sizes, int n_in,
                              void* d_out, int out_size, void* d_ws, size_t ws_size,
                              hipStream_t stream) {
  const float* x = (const float*)d_in[0];
  const int* mask = (const int*)d_in[1];
  const float* pos_emb = (const float*)d_in[2];
  const float* sim_p_w = (const float*)d_in[3];
  const float* sim_p_b = (const float*)d_in[4];
  const float* sim_n_w = (const float*)d_in[5];
  const float* sim_n_b = (const float*)d_in[6];
  const float* adj_w = (const float*)d_in[7];
  const float* gp1 = (const float*)d_in[8];
  const float* gp2 = (const float*)d_in[9];
  const float* gn1 = (const float*)d_in[10];
  const float* gn2 = (const float*)d_in[11];
  const float* ln1g = (const float*)d_in[12]; const float* ln1b = (const float*)d_in[13];
  const float* ln2g = (const float*)d_in[14]; const float* ln2b = (const float*)d_in[15];
  const float* ln3g = (const float*)d_in[16]; const float* ln3b = (const float*)d_in[17];
  const float* ln4g = (const float*)d_in[18]; const float* ln4b = (const float*)d_in[19];
  const float* ln5g = (const float*)d_in[20]; const float* ln5b = (const float*)d_in[21];
  const float* i2q_wq = (const float*)d_in[22];
  const float* i2q_wk = (const float*)d_in[23];
  const float* i2q_wv = (const float*)d_in[24];
  const float* i2q_wo = (const float*)d_in[25];
  const float* q2i_wq = (const float*)d_in[26];
  const float* q2i_wk = (const float*)d_in[27];
  const float* q2i_wv = (const float*)d_in[28];
  const float* q2i_wo = (const float*)d_in[29];
  const float* i2q_bo = (const float*)d_in[30];
  const float* q2i_bo = (const float*)d_in[31];
  const float* mlp_w = (const float*)d_in[32];
  const float* mlp_b = (const float*)d_in[33];
  const float* g_i2t = (const float*)d_in[34];
  const float* g_t2i = (const float*)d_in[35];

  float* out = (float*)d_out;                      // [B,N,D]
  float* out_psim = out + (long long)NB * SEQ * DIM;
  float* out_nsim = out_psim + (long long)NB * SEQ;

  // ---- workspace layout ----
  char* wsb = (char*)d_ws;
  size_t off = 0;
  auto wsf = [&](size_t n) { float* p = (float*)(wsb + off); off += ((n * 4 + 255) & ~(size_t)255); return p; };
  auto wsh = [&](size_t n) { _Float16* p = (_Float16*)(wsb + off); off += ((n * 2 + 255) & ~(size_t)255); return p; };
  auto wsi = [&](size_t n) { int* p = (int*)(wsb + off); off += ((n * 4 + 255) & ~(size_t)255); return p; };

  _Float16* W16[14];  // transposed fp16 weights: W16[i][n*DIM + k] = w[k*DIM + n]
  for (int i = 0; i < 14; ++i) W16[i] = wsh((size_t)DIM * DIM);
  float* pq = wsf(NB * DIM);
  float* nqv = wsf(NB * DIM);
  float* tp = wsf(NB * DIM);
  float* tn = wsf(NB * DIM);
  float* cp = wsf(NB);
  float* cn = wsf(NB);
  float* simsp = wsf((size_t)NB * SEQ);
  float* simsn = wsf((size_t)NB * SEQ);
  int* idxp = wsi(NB * TOPP);
  float* valp = wsf(NB * TOPP);
  int* idxn = wsi(NB * TOPN);
  float* valn = wsf(NB * TOPN);
  _Float16* nodep16 = wsh((size_t)NB * TOPP * DIM);
  _Float16* noden16 = wsh((size_t)NB * TOPN * DIM);
  _Float16* s1_16 = wsh((size_t)NB * TOPP * DIM);
  _Float16* h1_16 = wsh((size_t)NB * TOPP * DIM);   // stored transposed [DIM][K]
  _Float16* h16b = wsh((size_t)NB * TOPP * DIM);    // row-major (A operand)
  _Float16* h2_16 = wsh((size_t)NB * TOPP * DIM);   // stored transposed [DIM][K]
  _Float16* adj16 = wsh((size_t)NB * TOPP * TOPP);
  float* logits = wsf((size_t)NB * TOPP * TOPP);
  float* qgp = wsf((size_t)NB * TOPP * DIM);
  float* qgn = wsf((size_t)NB * TOPN * DIM);
  float* query = wsf((size_t)NB * NQT * DIM);
  float* query2 = wsf((size_t)NB * NQT * DIM);
  float* o1_32 = wsf((size_t)NB * NQT * DIM);
  _Float16* qln16 = wsh((size_t)NB * NQT * DIM);
  _Float16* kvln16 = wsh((size_t)NB * NQT * DIM);
  _Float16* qp16 = wsh((size_t)NB * NQT * DIM);
  _Float16* attn1_16 = wsh((size_t)NB * NQT * DIM);
  _Float16* k2_16 = wsh((size_t)NB * NQT * DIM);
  _Float16* v2t_16 = wsh((size_t)NB * NQT * DIM);   // transposed [DIM][NQT]
  _Float16* bufA16 = wsh((size_t)NB * SEQ * DIM);   // ln2 / ln3 / ln5 of feat
  _Float16* bufB16 = wsh((size_t)NB * SEQ * DIM);   // kproj -> q2
  _Float16* bufC16 = wsh((size_t)NB * SEQ * DIM);   // vprojT -> attn2 out
  float* buf32 = wsf((size_t)NB * SEQ * DIM);       // o2 -> mlp out

  auto gemm = [&](const _Float16* A, long long sA, const _Float16* Bt, long long sB,
                  float* C32, _Float16* C16, long long sC,
                  int M, int N, int K, int lda, int ldb, int transC16, int relu) {
    GemmP p;
    p.A = A; p.Bt = Bt; p.C32 = C32; p.C16 = C16;
    p.sA = sA; p.sB = sB; p.sC = sC;
    p.M = M; p.N = N; p.K = K; p.lda = lda; p.ldb = ldb;
    p.transC16 = transC16; p.relu = relu;
    dim3 g(M / 128, N / 64, NB);
    gemm_wmma<<<g, 128, 0, stream>>>(p);
  };

  // 1) weights -> fp16, TRANSPOSED to [N][K] for contiguous B fragments
  const float* wsrc[14] = {adj_w, gp1, gp2, gn1, gn2, i2q_wq, i2q_wk, i2q_wv,
                           i2q_wo, q2i_wq, q2i_wk, q2i_wv, q2i_wo, mlp_w};
  for (int i = 0; i < 14; ++i)
    f2h_t_kernel<<<(DIM * DIM) / 256, 256, 0, stream>>>(wsrc[i], W16[i]);

  // 2) masked means; 3) sim folded matvecs; 4) sims + sigmoid
  masked_mean_kernel<<<NB, DIM, 0, stream>>>(x, mask, pq, nqv);
  simvec_kernel<<<NB, DIM, 0, stream>>>(sim_p_w, sim_p_b, pq, tp, cp);
  simvec_kernel<<<NB, DIM, 0, stream>>>(sim_n_w, sim_n_b, nqv, tn, cn);
  sims_kernel<<<NB * SEQ / 8, 256, 0, stream>>>(x, tp, cp, tn, cn,
                                                out_psim, out_nsim, simsp, simsn);

  // 5) top-k ; 6) gather nodes (thresholded)
  topk_kernel<<<NB, 256, 0, stream>>>(simsp, idxp, valp, TOPP);
  topk_kernel<<<NB, 256, 0, stream>>>(simsn, idxn, valn, TOPN);
  gather_kernel<<<NB * TOPP, DIM, 0, stream>>>(x, idxp, valp, nodep16, TOPP);
  gather_kernel<<<NB * TOPN, DIM, 0, stream>>>(x, idxn, valn, noden16, TOPN);

  // 7) GCN (pos, K=TOPP).  node row-major [K,DIM] doubles as Bt for s1@node^T.
  gemm(nodep16, (long long)TOPP * DIM, W16[0], 0, nullptr, s1_16, (long long)TOPP * DIM,
       TOPP, DIM, DIM, DIM, DIM, 0, 0);
  gemm(s1_16, (long long)TOPP * DIM, nodep16, (long long)TOPP * DIM, logits, nullptr,
       (long long)TOPP * TOPP, TOPP, TOPP, DIM, DIM, DIM, 0, 0);
  softmax_kernel<<<NB * TOPP, 256, 0, stream>>>(logits, adj16, TOPP, 0.0625f);
  gemm(nodep16, (long long)TOPP * DIM, W16[1], 0, nullptr, h1_16, (long long)TOPP * DIM,
       TOPP, DIM, DIM, DIM, DIM, 1, 0);  // h1 stored transposed
  gemm(adj16, (long long)TOPP * TOPP, h1_16, (long long)TOPP * DIM, nullptr, h16b,
       (long long)TOPP * DIM, TOPP, DIM, TOPP, TOPP, TOPP, 0, 1);
  gemm(h16b, (long long)TOPP * DIM, W16[2], 0, nullptr, h2_16, (long long)TOPP * DIM,
       TOPP, DIM, DIM, DIM, DIM, 1, 0);  // h2 stored transposed
  gemm(adj16, (long long)TOPP * TOPP, h2_16, (long long)TOPP * DIM, qgp, nullptr,
       (long long)TOPP * DIM, TOPP, DIM, TOPP, TOPP, TOPP, 0, 0);

  // 8) GCN (neg, K=TOPN) reusing temps
  gemm(noden16, (long long)TOPN * DIM, W16[0], 0, nullptr, s1_16, (long long)TOPN * DIM,
       TOPN, DIM, DIM, DIM, DIM, 0, 0);
  gemm(s1_16, (long long)TOPN * DIM, noden16, (long long)TOPN * DIM, logits, nullptr,
       (long long)TOPN * TOPN, TOPN, TOPN, DIM, DIM, DIM, 0, 0);
  softmax_kernel<<<NB * TOPN, 256, 0, stream>>>(logits, adj16, TOPN, 0.0625f);
  gemm(noden16, (long long)TOPN * DIM, W16[3], 0, nullptr, h1_16, (long long)TOPN * DIM,
       TOPN, DIM, DIM, DIM, DIM, 1, 0);
  gemm(adj16, (long long)TOPN * TOPN, h1_16, (long long)TOPN * DIM, nullptr, h16b,
       (long long)TOPN * DIM, TOPN, DIM, TOPN, TOPN, TOPN, 0, 1);
  gemm(h16b, (long long)TOPN * DIM, W16[4], 0, nullptr, h2_16, (long long)TOPN * DIM,
       TOPN, DIM, DIM, DIM, DIM, 1, 0);
  gemm(adj16, (long long)TOPN * TOPN, h2_16, (long long)TOPN * DIM, qgn, nullptr,
       (long long)TOPN * DIM, TOPN, DIM, TOPN, TOPN, TOPN, 0, 0);

  // 9) query = concat(gcn_p + pe[ip], gcn_n + pe[in])
  build_query_kernel<<<NB * NQT, DIM, 0, stream>>>(qgp, qgn, idxp, idxn, pos_emb, query);

  // 10-14) i2q cross attention: query += g_i2t * CA(ln1(query), ln2(feat))
  ln_kernel<<<NB * NQT, DIM, 0, stream>>>(query, ln1g, ln1b, qln16);
  ln_kernel<<<NB * SEQ, DIM, 0, stream>>>(x, ln2g, ln2b, bufA16);
  gemm(qln16, (long long)NQT * DIM, W16[5], 0, nullptr, qp16, (long long)NQT * DIM,
       NQT, DIM, DIM, DIM, DIM, 0, 0);
  gemm(bufA16, (long long)SEQ * DIM, W16[6], 0, nullptr, bufB16, (long long)SEQ * DIM,
       SEQ, DIM, DIM, DIM, DIM, 0, 0);
  gemm(bufA16, (long long)SEQ * DIM, W16[7], 0, nullptr, bufC16, (long long)SEQ * DIM,
       SEQ, DIM, DIM, DIM, DIM, 1, 0);  // V stored transposed [DIM][SEQ]
  attn_kernel<<<dim3(NQT / 16, NH, NB), 32, 0, stream>>>(
      qp16, bufB16, bufC16, attn1_16, NQT, SEQ, SEQ,
      (long long)NQT * DIM, (long long)SEQ * DIM, (long long)SEQ * DIM,
      (long long)NQT * DIM, 0.125f);
  gemm(attn1_16, (long long)NQT * DIM, W16[8], 0, o1_32, nullptr, (long long)NQT * DIM,
       NQT, DIM, DIM, DIM, DIM, 0, 0);
  gate_res_kernel<<<NB * NQT, DIM, 0, stream>>>(query, o1_32, i2q_bo, g_i2t, query2);

  // 15-18) q2i cross attention: featb = x + g_t2i * CA(ln3(x), ln4(query2))
  ln_kernel<<<NB * SEQ, DIM, 0, stream>>>(x, ln3g, ln3b, bufA16);
  ln_kernel<<<NB * NQT, DIM, 0, stream>>>(query2, ln4g, ln4b, kvln16);
  gemm(bufA16, (long long)SEQ * DIM, W16[9], 0, nullptr, bufB16, (long long)SEQ * DIM,
       SEQ, DIM, DIM, DIM, DIM, 0, 0);
  gemm(kvln16, (long long)NQT * DIM, W16[10], 0, nullptr, k2_16, (long long)NQT * DIM,
       NQT, DIM, DIM, DIM, DIM, 0, 0);
  gemm(kvln16, (long long)NQT * DIM, W16[11], 0, nullptr, v2t_16, (long long)NQT * DIM,
       NQT, DIM, DIM, DIM, DIM, 1, 0);  // V stored transposed [DIM][NQT]
  attn_kernel<<<dim3(SEQ / 16, NH, NB), 32, 0, stream>>>(
      bufB16, k2_16, v2t_16, bufC16, SEQ, NQT, NQT,
      (long long)SEQ * DIM, (long long)NQT * DIM, (long long)NQT * DIM,
      (long long)SEQ * DIM, 0.125f);
  gemm(bufC16, (long long)SEQ * DIM, W16[12], 0, buf32, nullptr, (long long)SEQ * DIM,
       SEQ, DIM, DIM, DIM, DIM, 0, 0);
  gate_res_kernel<<<NB * SEQ, DIM, 0, stream>>>(x, buf32, q2i_bo, g_t2i, out);

  // 19-21) MLP: out += ln5(out) @ mlp_w + mlp_b
  ln_kernel<<<NB * SEQ, DIM, 0, stream>>>(out, ln5g, ln5b, bufA16);
  gemm(bufA16, (long long)SEQ * DIM, W16[13], 0, buf32, nullptr, (long long)SEQ * DIM,
       SEQ, DIM, DIM, DIM, DIM, 0, 0);
  final_add_kernel<<<NB * SEQ, DIM, 0, stream>>>(out, buf32, mlp_b);
}
